// SelfAttention2D_65481071396253
// MI455X (gfx1250) — matrix operand order repaired and checked by reading, not compiled, not run on hardware
//
#include <hip/hip_runtime.h>
#include <stdint.h>

typedef __attribute__((ext_vector_type(16))) _Float16 v16h;
typedef __attribute__((ext_vector_type(8)))  float    v8f;

union HOp { uint32_t u[8]; v16h v; };

__device__ inline uint32_t pack_f16(float lo, float hi) {
    union { _Float16 h[2]; uint32_t u; } t;
    t.h[0] = (_Float16)lo;
    t.h[1] = (_Float16)hi;
    return t.u;
}
__device__ inline uint16_t f32_f16(float f) {
    union { _Float16 h; uint16_t u; } t;
    t.h = (_Float16)f;
    return t.u;
}

// A-operand (16x32 f16, M x K) K index for vgpr-pair j, lane-half h
__device__ inline int aK(int j, int h) { return 16 * (j >> 2) + 2 * (j & 3) + 8 * h; }
// B-operand (32x16 f16, K x N) K index for vgpr-pair j, lane-half h
__device__ inline int bK(int j, int h) { return 16 * (j >> 2) + 2 * (j & 3) + 8 * h; }

#define WMMA_F16(A, B, Cc) \
    __builtin_amdgcn_wmma_f32_16x16x32_f16(false, (A), false, (B), (short)0, (Cc), false, false)

// ---------------------------------------------------------------------------
// Kernel 1: fused q/k/v projections.  O[n][o] = sum_c x[c][n] * W[o][c] + b[o]
// o in [0,320): 0-31 -> q, 32-63 -> k, 64-319 -> v.
// q stored [b][n][32] f16, k stored [b][m][32] f16, v stored [b][c][m] f16.
// One wave per (batch, 16-query tile); 20 o-tiles x 8 K-chunks of WMMA.
// ---------------------------------------------------------------------------
__global__ __launch_bounds__(128) void proj_kernel(
    const float* __restrict__ x,
    const float* __restrict__ Wq, const float* __restrict__ bq,
    const float* __restrict__ Wk, const float* __restrict__ bk,
    const float* __restrict__ Wv, const float* __restrict__ bv,
    uint16_t* __restrict__ qws, uint16_t* __restrict__ kws,
    uint16_t* __restrict__ vws)
{
    const int lane = threadIdx.x & 31;
    const int wave = threadIdx.x >> 5;
    const int col  = lane & 15;
    const int half = lane >> 4;
    const int tile = blockIdx.x * 4 + wave;   // 0..1023
    const int b    = tile >> 8;
    const int n0   = (tile & 255) << 4;

    // A operands: x^T tiles (16 n x 32 c), 8 chunks cover K=C=256.
    HOp XA[8];
#pragma unroll
    for (int kc = 0; kc < 8; ++kc) {
#pragma unroll
        for (int j = 0; j < 8; ++j) {
            int c = kc * 32 + aK(j, half);
            const float* p = x + (size_t)(b * 256 + c) * 4096 + n0 + col;
            XA[kc].u[j] = pack_f16(p[0], p[4096]);   // (c, c+1)
        }
    }

#pragma unroll 1
    for (int ot = 0; ot < 20; ++ot) {
        const int o0    = ot << 4;
        const int olane = o0 + col;
        const float* Wm; const float* bias; int row;
        if (o0 < 32)      { Wm = Wq; bias = bq; row = olane; }
        else if (o0 < 64) { Wm = Wk; bias = bk; row = olane - 32; }
        else              { Wm = Wv; bias = bv; row = olane - 64; }

        v8f acc = {};
#pragma unroll
        for (int kc = 0; kc < 8; ++kc) {
            HOp WB;   // B operand: W^T (c x o), column (fixed o) = W row, contiguous
#pragma unroll
            for (int j = 0; j < 8; ++j) {
                int c = kc * 32 + bK(j, half);
                const float* p = Wm + row * 256 + c;
                WB.u[j] = pack_f16(p[0], p[1]);
            }
            acc = WMMA_F16(XA[kc].v, WB.v, acc);
        }
        const float bb = bias[row];
#pragma unroll
        for (int r = 0; r < 8; ++r) {
            const int n = n0 + r + 8 * half;           // D: M=n over vgprs
            const uint16_t hval = f32_f16(acc[r] + bb);
            if (o0 < 32)
                qws[(size_t)(b * 4096 + n) * 32 + olane] = hval;
            else if (o0 < 64)
                kws[(size_t)(b * 4096 + n) * 32 + (olane - 32)] = hval;
            else
                vws[(size_t)(b * 256 + (olane - 64)) * 4096 + n] = hval;
        }
    }
}

// ---------------------------------------------------------------------------
// Kernel 2: fused flash attention + residual.
// Block = 128 thr (4 waves) = 2 query tiles x 2 channel halves.  Each wave:
// 16 queries x 128 channels (8 f32 C/D tiles, 64 acc VGPRs), so the energy
// WMMAs + softmax VALU tail are only 2x redundant per query tile.
// Per 32-key step: E^T = K_tile x Q^T (2 WMMA), online softmax in the D
// layout (lane keyed by query n), P (exp'd, f16-packed in place) feeds the
// 8 P.V WMMAs directly as the A operand -- no cross-lane transpose needed.
// ---------------------------------------------------------------------------
__global__ __launch_bounds__(128) void attn_kernel(
    const uint16_t* __restrict__ qws, const uint16_t* __restrict__ kws,
    const uint16_t* __restrict__ vws, const float* __restrict__ x,
    const float* __restrict__ gamma, float* __restrict__ out)
{
    const int lane = threadIdx.x & 31;
    const int wave = threadIdx.x >> 5;
    const int col  = lane & 15;
    const int half = lane >> 4;
    const int gw   = blockIdx.x * 4 + wave;   // 0..2047
    const int qt   = gw >> 1;                 // 0..1023: (batch, query tile)
    const int b    = qt >> 8;
    const int n0   = (qt & 255) << 4;
    const int cb   = (gw & 1) << 7;           // 128-channel half

    const uint16_t* qb = qws + (size_t)b * 4096 * 32;
    const uint16_t* kb = kws + (size_t)b * 4096 * 32;
    const uint16_t* vb = vws + (size_t)b * 256 * 4096;

    // Q^T as B operand (o x n): column (fixed n) = q row, contiguous.
    HOp QB;
#pragma unroll
    for (int j = 0; j < 8; ++j)
        QB.u[j] = *(const uint32_t*)(qb + (size_t)(n0 + col) * 32 + bK(j, half));

    float m_run = -3.0e38f;
    float l_run = 0.0f;
    v8f acc[8];
#pragma unroll
    for (int t = 0; t < 8; ++t) acc[t] = (v8f){};

    for (int m0 = 0; m0 < 4096; m0 += 32) {
        // K rows as A operands (m x o), rows contiguous.
        HOp KA0, KA1;
#pragma unroll
        for (int j = 0; j < 8; ++j) {
            const int o = aK(j, half);
            KA0.u[j] = *(const uint32_t*)(kb + (size_t)(m0 + col) * 32 + o);
            KA1.u[j] = *(const uint32_t*)(kb + (size_t)(m0 + 16 + col) * 32 + o);
        }
        if (m0 + 32 < 4096)
            __builtin_prefetch(kb + (size_t)(m0 + 32 + col) * 32, 0, 3);

        v8f e0 = {}, e1 = {};
        e0 = WMMA_F16(KA0.v, QB.v, e0);   // E^T[m0..m0+15][n]
        e1 = WMMA_F16(KA1.v, QB.v, e1);   // E^T[m0+16..m0+31][n]

        // Online softmax.  Row n = col lives in 2 lanes (both halves).
        float tmax = e0[0];
#pragma unroll
        for (int r = 1; r < 8; ++r) tmax = fmaxf(tmax, e0[r]);
#pragma unroll
        for (int r = 0; r < 8; ++r) tmax = fmaxf(tmax, e1[r]);
        tmax = fmaxf(tmax, __shfl_xor(tmax, 16, 32));
        const float m_new = fmaxf(m_run, tmax);
        const float scale = __expf(m_run - m_new);

        float p0[8], p1[8], rsum = 0.0f;
#pragma unroll
        for (int r = 0; r < 8; ++r) {
            p0[r] = __expf(e0[r] - m_new);
            p1[r] = __expf(e1[r] - m_new);
            rsum += p0[r] + p1[r];
        }
        rsum += __shfl_xor(rsum, 16, 32);
        l_run = l_run * scale + rsum;
        m_run = m_new;

        // Rescale accumulators: row n = r + 8*half -> broadcast scale(n).
#pragma unroll
        for (int r = 0; r < 8; ++r) {
            const float sr = __int_as_float(__builtin_amdgcn_ds_bpermute(
                (r + 8 * half) * 4, __float_as_int(scale)));
#pragma unroll
            for (int t = 0; t < 8; ++t) acc[t][r] *= sr;
        }

        // P in D-layout == A-operand layout: pure per-lane f16 pack.
        HOp PA;
#pragma unroll
        for (int j = 0; j < 4; ++j) PA.u[j]     = pack_f16(p0[2 * j], p0[2 * j + 1]);
#pragma unroll
        for (int j = 0; j < 4; ++j) PA.u[4 + j] = pack_f16(p1[2 * j], p1[2 * j + 1]);

        // out[n][c] += P[n][m] * V[m][c]; V stored [c][m] -> B columns contiguous.
#pragma unroll
        for (int t = 0; t < 8; ++t) {
            HOp VB;
#pragma unroll
            for (int j = 0; j < 8; ++j)
                VB.u[j] = *(const uint32_t*)(vb + (size_t)(cb + 16 * t + col) * 4096
                                             + m0 + bK(j, half));
            if (t == 0 && m0 + 32 < 4096)
                __builtin_prefetch(vb + (size_t)(cb + col) * 4096 + m0 + 32, 0, 3);
            acc[t] = WMMA_F16(PA.v, VB.v, acc[t]);
        }
    }

    // Epilogue: out = gamma * (acc / l) + x
    const float inv = 1.0f / l_run;
    const float g   = gamma[0];
#pragma unroll
    for (int r = 0; r < 8; ++r) {
        const float ir = __int_as_float(__builtin_amdgcn_ds_bpermute(
            (r + 8 * half) * 4, __float_as_int(inv)));
        const int n = n0 + r + 8 * half;
#pragma unroll
        for (int t = 0; t < 8; ++t) {
            const size_t idx = (size_t)(b * 256 + cb + 16 * t + col) * 4096 + n;
            out[idx] = g * (acc[t][r] * ir) + x[idx];
        }
    }
}

extern "C" void kernel_launch(void* const* d_in, const int* in_sizes, int n_in,
                              void* d_out, int out_size, void* d_ws, size_t ws_size,
                              hipStream_t stream) {
    const float* x  = (const float*)d_in[0];
    const float* Wq = (const float*)d_in[1];
    const float* bq = (const float*)d_in[2];
    const float* Wk = (const float*)d_in[3];
    const float* bk = (const float*)d_in[4];
    const float* Wv = (const float*)d_in[5];
    const float* bv = (const float*)d_in[6];
    const float* gm = (const float*)d_in[7];
    float* out = (float*)d_out;

    uint16_t* qws = (uint16_t*)d_ws;                    // 4*4096*32 f16 = 1 MB
    uint16_t* kws = qws + (size_t)4 * 4096 * 32;        // 1 MB
    uint16_t* vws = kws + (size_t)4 * 4096 * 32;        // 4*256*4096 f16 = 8 MB

    proj_kernel<<<256, 128, 0, stream>>>(x, Wq, bq, Wk, bk, Wv, bv, qws, kws, vws);
    attn_kernel<<<512, 128, 0, stream>>>(qws, kws, vws, x, gm, out);
}
